// VectorQuantizer_90194313216180
// MI455X (gfx1250) — matrix operand order repaired and checked by reading, not compiled, not run on hardware
//
#include <hip/hip_runtime.h>
#include <hip/hip_bf16.h>

// ---------------------------------------------------------------------------
// VQ-VAE vector quantizer for MI455X (gfx1250, wave32, WMMA).
//   z_e:       (16, 256, 64, 64) f32   -> 65536 query rows of dim 256
//   embedding: (1024, 256) f32
//   out:       z_q (16,256,64,64) f32  ++  scalar vq_loss
// Distance GEMM via v_wmma_f32_16x16x32_f16 with the codebook tile staged in
// LDS (shared by the 8 waves of the workgroup -> 8x less L2 traffic), argmin
// reduced in-register.  Codebook tiles are double-buffered with gfx1250
// async global->LDS DMA (global_load_async_to_lds_b128 / s_wait_asynccnt).
// ---------------------------------------------------------------------------

typedef __attribute__((ext_vector_type(16))) _Float16 v16h;
typedef __attribute__((ext_vector_type(8)))  _Float16 v8h;
typedef __attribute__((ext_vector_type(8)))  float    v8f;

#define N_EMBED 1024
#define EMBED_C 256
#define NB      16
#define HW      4096            // 64*64
#define NPOS    65536           // 16*64*64
#define OUT_ELEMS 16777216      // 16*256*64*64

// Padded LDS row: 264 halves = 132 dwords; 132 % 64 == 4 -> the 16 lanes of a
// half-wave (stride 4 banks, 4-dword b128 reads) cover all 64 banks with no
// conflict, and 264*2 bytes keeps every fragment 16-byte aligned.
#define LROW 264

#if defined(__HIP_DEVICE_COMPILE__) && \
    __has_builtin(__builtin_amdgcn_global_load_async_to_lds_b128) && \
    __has_builtin(__builtin_amdgcn_s_wait_asynccnt)
#define USE_ASYNC_LDS 1
#define AS1 __attribute__((address_space(1)))
#define AS3 __attribute__((address_space(3)))
// exact pointee type expected by the builtin (per hipcc diagnostic):
typedef int v4i_async __attribute__((vector_size(4 * sizeof(int))));
#else
#define USE_ASYNC_LDS 0
#endif

// --------------------------------------------------------------------------
// Kernel 0: zero the scalar loss accumulator (last element of d_out).
// --------------------------------------------------------------------------
__global__ void vq_zero_loss_kernel(float* loss) {
    if (threadIdx.x == 0 && blockIdx.x == 0) *loss = 0.0f;
}

// --------------------------------------------------------------------------
// Kernel 1: embedding prep.  One block per code row: f32 -> f16 copy and
// ||e_k||^2 reduction.  eh (512 KB) stays L2-resident afterwards.
// --------------------------------------------------------------------------
__global__ __launch_bounds__(EMBED_C) void vq_emb_prep_kernel(
    const float* __restrict__ emb, _Float16* __restrict__ eh,
    float* __restrict__ norms) {
    __shared__ float red[EMBED_C];
    const int k = blockIdx.x;
    const int c = threadIdx.x;
    const float v = emb[(size_t)k * EMBED_C + c];
    eh[(size_t)k * EMBED_C + c] = (_Float16)v;
    red[c] = v * v;
    __syncthreads();
    #pragma unroll
    for (int s = EMBED_C / 2; s > 0; s >>= 1) {
        if (c < s) red[c] += red[c + s];
        __syncthreads();
    }
    if (c == 0) norms[k] = red[0];
}

// --------------------------------------------------------------------------
// Kernel 2: LDS-tiled transpose+convert  z_e (B,C,H,W) -> zh (N=B*HW, C) f16.
// 32x32 f32 tiles; both global read (along hw) and write (along c) coalesced.
// --------------------------------------------------------------------------
__global__ __launch_bounds__(1024) void vq_z_prep_kernel(
    const float* __restrict__ z, _Float16* __restrict__ zh) {
    __shared__ float tile[32][33];
    const int hw0 = blockIdx.x * 32;
    const int c0  = blockIdx.y * 32;
    const int b   = blockIdx.z;
    const int tx = threadIdx.x & 31;
    const int ty = threadIdx.x >> 5;
    tile[ty][tx] = z[((size_t)b * EMBED_C + (c0 + ty)) * HW + hw0 + tx];
    __syncthreads();
    zh[((size_t)(b * HW + hw0 + ty)) * EMBED_C + c0 + tx] = (_Float16)tile[tx][ty];
}

// --------------------------------------------------------------------------
// Kernel 3: WMMA distance + argmin with LDS-staged codebook tiles.
//
// Workgroup = 8 waves, 128 rows.  Each wave owns 16 rows: A-fragments
// (16 x K=256 f16 = 64 VGPRs) loaded once.  The 64 codebook tiles
// (16 codes x 256 f16 = 8 KB each) are staged into a double-buffered LDS
// tile by the whole workgroup (async DMA one tile ahead), then every wave
// runs 8 v_wmma_f32_16x16x32_f16 per tile out of LDS on two independent
// accumulator chains.  dist = ||e||^2 - 2 x.e; running per-lane
// (min,argmin), final 16-lane shfl_xor reduce.
//
// ISA fragment layouts (CDNA5 ISA):
//   A 16x32 f16: lane<16 -> row=lane, v0..3=K0..7, v4..7=K16..23;
//                lane>=16 -> same row, K8..15 / K24..31.
//   B 32x16 f16 (as loaded here, A likewise): lane<16 -> col=lane, v0..7=K0..15 (contig);
//                lane>=16 -> K16..31.
//   C 16x16 f32: VGPR r: lanes0-15 -> row r, N=lane; lanes16-31 -> row r+8.
// --------------------------------------------------------------------------
__global__ __launch_bounds__(256) void vq_argmin_kernel(
    const _Float16* __restrict__ zh,     // [NPOS, 256]
    const _Float16* __restrict__ eh,     // [1024, 256]
    const float* __restrict__ norms,     // [1024]
    int* __restrict__ indices) {         // [NPOS]
    __shared__ __align__(16) _Float16 lbuf[2][16][LROW];

    const int tid  = threadIdx.x;
    const int lane = tid & 31;
    const int wave = blockIdx.x * 8 + (tid >> 5);
    const int n0   = wave * 16;
    const int l16  = lane & 15;
    const int hi   = lane >> 4;          // which half of the wave

    // stage one 16-code tile (8 KB) into LDS buffer b: 256 threads x 32 B
    const int srow   = tid >> 4;         // 0..15 code row within tile
    const int schunk = (tid & 15) * 16;  // 16-half chunk within row
    auto stage = [&](int t, int b) {
        const _Float16* g = eh + ((size_t)(t * 16 + srow)) * EMBED_C + schunk;
        _Float16* l = &lbuf[b][srow][schunk];
#if USE_ASYNC_LDS
        __builtin_amdgcn_global_load_async_to_lds_b128(
            (AS1 v4i_async*)g, (AS3 v4i_async*)l, 0, 0);
        __builtin_amdgcn_global_load_async_to_lds_b128(
            (AS1 v4i_async*)(g + 8), (AS3 v4i_async*)(l + 8), 0, 0);
#else
        *(v8h*)l       = *(const v8h*)g;
        *(v8h*)(l + 8) = *(const v8h*)(g + 8);
#endif
    };

    // ---- A fragments: 16 rows x K=256, resident for the whole sweep ----
    const _Float16* arow = zh + (size_t)(n0 + l16) * EMBED_C;
    v16h a[8];
    #pragma unroll
    for (int f = 0; f < 8; ++f) {
        const _Float16* p = arow + f * 32 + hi * 16;
        union { v16h v; v8h h[2]; } u;
        u.h[0] = *(const v8h*)(p);        // K f*32 + (hi?16:0)  .. +7
        u.h[1] = *(const v8h*)(p + 8);    // K f*32 + 8 + (hi?16:0) .. +7
        a[f] = u.v;
    }

    float minv[8];
    int   mini[8];
    #pragma unroll
    for (int r = 0; r < 8; ++r) { minv[r] = 3.4e38f; mini[r] = 0; }

    stage(0, 0);
    for (int t = 0; t < 64; ++t) {
        const int cur = t & 1;
#if USE_ASYNC_LDS
        __builtin_amdgcn_s_wait_asynccnt(0);   // tile t landed in LDS
#endif
        __syncthreads();                       // publish tile t; retire reads
                                               // of the buffer stage() hits
        if (t < 63) stage(t + 1, cur ^ 1);

        const _Float16* lrow = lbuf[cur][l16]; // this lane's code column
        v8f acc0 = {}, acc1 = {};              // two independent XDL chains
        #pragma unroll
        for (int f = 0; f < 8; f += 2) {
            v16h b0 = *(const v16h*)(lrow + f * 32 + hi * 16);
            v16h b1 = *(const v16h*)(lrow + (f + 1) * 32 + hi * 16);
            acc0 = __builtin_amdgcn_wmma_f32_16x16x32_f16(
                false, a[f], false, b0, (short)0, acc0, false, false);
            acc1 = __builtin_amdgcn_wmma_f32_16x16x32_f16(
                false, a[f + 1], false, b1, (short)0, acc1, false, false);
        }

        const int col = t * 16 + l16;
        const float nrm = norms[col];
        #pragma unroll
        for (int r = 0; r < 8; ++r) {
            const float d = nrm - 2.0f * (acc0[r] + acc1[r]); // ||e||^2 - 2 x.e
            if (d < minv[r]) { minv[r] = d; mini[r] = col; }
        }
    }

    // ---- min+argmin across the 16 lanes of each half (xor of bits 0..3
    //      never crosses the half boundary) ----
    #pragma unroll
    for (int m = 8; m >= 1; m >>= 1) {
        #pragma unroll
        for (int r = 0; r < 8; ++r) {
            const float ov = __shfl_xor(minv[r], m, 32);
            const int   oi = __shfl_xor(mini[r], m, 32);
            if (ov < minv[r] || (ov == minv[r] && oi < mini[r])) {
                minv[r] = ov; mini[r] = oi;
            }
        }
    }

    if (l16 == 0) {
        const int rowbase = n0 + hi * 8;       // lane0 -> rows 0..7, lane16 -> 8..15
        #pragma unroll
        for (int r = 0; r < 8; ++r) indices[rowbase + r] = mini[r];
    }
}

// --------------------------------------------------------------------------
// Kernel 4: straight-through output + commitment loss.
// Forward value of z_q_st == embedding[argmin]; coalesced BCHW stores,
// codebook rows gathered from L2.  Loss reduced per block in LDS, one
// atomicAdd per block, pre-scaled by BETA/numel.
// --------------------------------------------------------------------------
__global__ __launch_bounds__(256) void vq_output_kernel(
    const float* __restrict__ z,          // original z_e, BCHW flat
    const float* __restrict__ emb,        // f32 codebook
    const int*   __restrict__ indices,    // [NPOS]
    float* __restrict__ out,              // z_q_st, BCHW flat
    float* __restrict__ loss) {
    __shared__ float red[256];
    const size_t o = (size_t)blockIdx.x * 256 + threadIdx.x;  // < OUT_ELEMS
    const int hw = (int)(o & 4095);
    const int c  = (int)((o >> 12) & 255);
    const int b  = (int)(o >> 20);
    const int n  = (b << 12) | hw;
    const int idx = indices[n];
    const float q  = emb[(size_t)idx * EMBED_C + c];
    const float ze = z[o];
    out[o] = q;                                      // z_e + sg(z_q - z_e) == z_q
    const float d = q - ze;
    red[threadIdx.x] = d * d;
    __syncthreads();
    #pragma unroll
    for (int s = 128; s > 0; s >>= 1) {
        if (threadIdx.x < s) red[threadIdx.x] += red[threadIdx.x + s];
        __syncthreads();
    }
    if (threadIdx.x == 0)
        atomicAdd(loss, red[0] * (0.25f / (float)OUT_ELEMS));
}

// --------------------------------------------------------------------------
extern "C" void kernel_launch(void* const* d_in, const int* in_sizes, int n_in,
                              void* d_out, int out_size, void* d_ws, size_t ws_size,
                              hipStream_t stream) {
    (void)in_sizes; (void)n_in; (void)out_size; (void)ws_size;
    const float* z_e = (const float*)d_in[0];   // 16*256*64*64 f32
    const float* emb = (const float*)d_in[1];   // 1024*256 f32
    float* out  = (float*)d_out;                // OUT_ELEMS z_q + 1 loss
    float* loss = out + (size_t)OUT_ELEMS;

    // workspace carve-up (~33 MB)
    char* ws = (char*)d_ws;
    _Float16* zh = (_Float16*)ws;  ws += (size_t)NPOS * EMBED_C * sizeof(_Float16);   // 32 MB
    _Float16* eh = (_Float16*)ws;  ws += (size_t)N_EMBED * EMBED_C * sizeof(_Float16);// 512 KB
    float* norms = (float*)ws;     ws += (size_t)N_EMBED * sizeof(float);             // 4 KB
    int* indices = (int*)ws;                                                          // 256 KB

    vq_zero_loss_kernel<<<1, 32, 0, stream>>>(loss);
    vq_emb_prep_kernel<<<N_EMBED, EMBED_C, 0, stream>>>(emb, eh, norms);
    vq_z_prep_kernel<<<dim3(HW / 32, EMBED_C / 32, NB), 1024, 0, stream>>>(z_e, zh);
    vq_argmin_kernel<<<NPOS / (16 * 8), 256, 0, stream>>>(zh, eh, norms, indices);
    vq_output_kernel<<<OUT_ELEMS / 256, 256, 0, stream>>>(z_e, emb, indices, out, loss);
}
